// MoE_13125420057043
// MI455X (gfx1250) — matrix operand order repaired and checked by reading, not compiled, not run on hardware
//
#include <hip/hip_runtime.h>
#include <cmath>

// ---------------- problem constants (match reference) ----------------
#define NUM_EXPERTS 32
#define INPUT_DIM   1024
#define K_TOP       8
#define BATCH       4096

// ---------------- GEMM tiling ----------------
#define BM 64            // tokens per block
#define BN 128           // N-tile (hidden cols / output cols)
#define BK 32            // K-step (one bf16 WMMA K)
#define STG_LD 40        // staged-tile row stride in bf16 elems (32 + 8 pad = 80B, 16B aligned)
#define SMAX  1536       // largest expert hidden size
#define SHMAX (SMAX + 8) // h row stride upper bound (bf16 elems)
#define H_ELEMS_MAX (BM * SHMAX)
#define A_BUF_ELEMS (BM * STG_LD)     // 2560
#define B_BUF_ELEMS (BN * STG_LD)     // 5120

static constexpr float BN_RSQRT = 0.9999950000374997f; // 1/sqrt(1 + 1e-5): eval BN scale

typedef __attribute__((ext_vector_type(2)))  __bf16 v2bf;
typedef __attribute__((ext_vector_type(8)))  __bf16 v8bf;
typedef __attribute__((ext_vector_type(16))) __bf16 v16bf;
typedef __attribute__((ext_vector_type(8)))  float  v8f;

struct ExpertParams {
  const float *w1, *b1, *g1, *be1, *w2, *b2, *g2, *be2;
};
struct AllExperts { ExpertParams e[NUM_EXPERTS]; };   // 2 KB kernarg, under 4 KB limit

// dynamic LDS: h ++ 2x Astage ++ 2x Bstage ++ tok[BM] ++ gate[BM]
static constexpr size_t FFN_SMEM =
    (size_t)(H_ELEMS_MAX + 2 * A_BUF_ELEMS + 2 * B_BUF_ELEMS) * sizeof(__bf16)
    + BM * sizeof(int) + BM * sizeof(float);          // 228864 B < 320 KB WGP LDS

// expert hidden size (same formula as reference, deterministic)
__device__ __forceinline__ int expert_size(int e) {
  return (int)(1024.0 * (0.5 + (double)e / 31.0));
}

// ---- CDNA5 async global->LDS copy (ASYNCcnt) + wait ----
__device__ __forceinline__ void async_copy16(void* lds_dst, const void* gsrc) {
  unsigned lds = (unsigned)(unsigned long long)lds_dst;  // low 32 bits = LDS offset
  unsigned long long g = (unsigned long long)gsrc;
  asm volatile("global_load_async_to_lds_b128 %0, %1, off"
               :: "v"(lds), "v"(g) : "memory");
}
__device__ __forceinline__ void wait_async0() {
#if __has_builtin(__builtin_amdgcn_s_wait_asynccnt)
  __builtin_amdgcn_s_wait_asynccnt(0);
#else
  asm volatile("s_wait_asynccnt 0x0" ::: "memory");
#endif
}

// pack two floats -> bf16x2 (single v_cvt_pk_bf16_f32)
__device__ __forceinline__ unsigned pack_bf16(float v0, float v1) {
  v2bf p = { (__bf16)v0, (__bf16)v1 };
  return __builtin_bit_cast(unsigned, p);
}

// ---- WMMA fragment loaders (ISA 7.12.2 wave32 layouts, 16B-aligned ds_load_b128) ----
// A 16x32 bf16: lane<16 (M=lane) holds K = {0..7, 16..23}; lane>=16 holds K = {8..15, 24..31}
__device__ __forceinline__ v16bf load_frag_a(const __bf16* base, int ld, int lane) {
  const int r = lane & 15, hf = lane >> 4;
  const __bf16* p = base + r * ld + hf * 8;
  v8bf lo = *(const v8bf*)(p);
  v8bf hi = *(const v8bf*)(p + 16);
  v16bf f;
#pragma unroll
  for (int i = 0; i < 8; ++i) { f[i] = lo[i]; f[8 + i] = hi[i]; }
  return f;
}
// B 32x16 bf16 staged K-contiguous as [N][K]: lane<16 (N=lane) holds K = {0..7, 16..23}; lane>=16 holds K = {8..15, 24..31}
__device__ __forceinline__ v16bf load_frag_b(const __bf16* base, int ld, int lane) {
  const int r = lane & 15, hf = lane >> 4;
  const __bf16* p = base + r * ld + hf * 8;
  v8bf lo = *(const v8bf*)(p);
  v8bf hi = *(const v8bf*)(p + 16);
  v16bf f;
#pragma unroll
  for (int i = 0; i < 8; ++i) { f[i] = lo[i]; f[8 + i] = hi[i]; }
  return f;
}

// ================= K0: out = x (residual) ; xb = bf16(x) =================
__global__ void moe_init_kernel(const float* __restrict__ x, float* __restrict__ out,
                                __bf16* __restrict__ xb) {
  int i = blockIdx.x * 256 + threadIdx.x;        // B*D total
  float v = x[i];
  out[i] = v;
  xb[i] = (__bf16)v;
}

// ================= K1: router logits + top-8 + softmax (lane-per-expert) =================
__global__ void moe_router_kernel(const float* __restrict__ x,
                                  const float* __restrict__ rw,   // [D][E]
                                  const float* __restrict__ rb,   // [E]
                                  float* __restrict__ gdense) {   // [B][E]
  const int lane = threadIdx.x & 31;
  const int wid  = threadIdx.x >> 5;
  const int t    = blockIdx.x * 8 + wid;          // one wave32 per token, lane = expert
  const float4* x4 = (const float4*)(x + (size_t)t * INPUT_DIM);
  float acc = rb[lane];
#pragma unroll 4
  for (int k4 = 0; k4 < INPUT_DIM / 4; ++k4) {
    float4 xv = x4[k4];                            // uniform across lanes
    int k = k4 * 4;
    acc = fmaf(xv.x, rw[(k + 0) * NUM_EXPERTS + lane], acc);
    acc = fmaf(xv.y, rw[(k + 1) * NUM_EXPERTS + lane], acc);
    acc = fmaf(xv.z, rw[(k + 2) * NUM_EXPERTS + lane], acc);
    acc = fmaf(xv.w, rw[(k + 3) * NUM_EXPERTS + lane], acc);
  }
  // top-8 by repeated wave argmax
  float val = acc, vmax = 0.f, myexp = 0.f;
  bool selected = false;
#pragma unroll
  for (int s = 0; s < K_TOP; ++s) {
    float m = val; int mi = lane;
#pragma unroll
    for (int off = 16; off > 0; off >>= 1) {
      float ov = __shfl_xor(m, off, 32);
      int   oi = __shfl_xor(mi, off, 32);
      if (ov > m || (ov == m && oi < mi)) { m = ov; mi = oi; }
    }
    if (s == 0) vmax = m;
    if (lane == mi) { selected = true; myexp = __expf(val - vmax); val = -1e30f; }
  }
  float tot = selected ? myexp : 0.f;
#pragma unroll
  for (int off = 16; off > 0; off >>= 1) tot += __shfl_xor(tot, off, 32);
  gdense[(size_t)t * NUM_EXPERTS + lane] = selected ? (myexp / tot) : 0.f;
}

// ================= K2: deterministic per-expert compaction =================
__global__ void moe_compact_kernel(const float* __restrict__ gdense,
                                   int* __restrict__ tok_ids, float* __restrict__ gates,
                                   int* __restrict__ counts) {
  const int e = blockIdx.x, tid = threadIdx.x, lane = tid & 31, wid = tid >> 5;
  __shared__ int base, wsum[8], woff[8];
  if (tid == 0) base = 0;
  __syncthreads();
  for (int t0 = 0; t0 < BATCH; t0 += 256) {
    int t = t0 + tid;
    float g = gdense[(size_t)t * NUM_EXPERTS + e];
    bool sel = (g != 0.0f);
    unsigned m = (unsigned)__ballot(sel);          // wave32: low 32 bits
    int prefix = __popc(m & ((1u << lane) - 1u));
    if (lane == 0) wsum[wid] = __popc(m);
    __syncthreads();
    if (tid == 0) {
      int a = base;
#pragma unroll
      for (int w = 0; w < 8; ++w) { woff[w] = a; a += wsum[w]; }
      base = a;
    }
    __syncthreads();
    if (sel) {
      int pos = woff[wid] + prefix;
      tok_ids[(size_t)e * BATCH + pos] = t;
      gates[(size_t)e * BATCH + pos]   = g;
    }
    __syncthreads();
  }
  if (tid == 0) counts[e] = base;
}

// ================= K3: fused 2-GEMM expert FFN, h in LDS, bf16 WMMA =================
// double-buffered staging, async A-tile DMA, one barrier per K-step
__global__ __launch_bounds__(256, 1)
void moe_ffn_kernel(AllExperts P, const __bf16* __restrict__ xb,
                    const int* __restrict__ tok_ids, const float* __restrict__ gates,
                    const int* __restrict__ counts, float* __restrict__ out) {
  extern __shared__ char smem[];
  const int bx = blockIdx.x;
  const int e = bx >> 6;               // 64 m-tiles per expert
  const int mtile = bx & 63;
  const int cnt = counts[e];
  const int mbase = mtile * BM;
  if (mbase >= cnt) return;

  const int S  = expert_size(e);
  const int Sc = (S + 31) & ~31;       // K padded to BK
  const int SH = Sc + 8;               // h row stride (16B aligned)

  __bf16* h_lds = (__bf16*)smem;                       // [BM][SH]
  __bf16* A_lds = ((__bf16*)smem) + H_ELEMS_MAX;       // [2][BM][STG_LD]
  __bf16* B_lds = A_lds + 2 * A_BUF_ELEMS;             // [2][BN][STG_LD] ([n][k])
  int*    tok_lds  = (int*)(B_lds + 2 * B_BUF_ELEMS);  // [BM]
  float*  gate_lds = (float*)(tok_lds + BM);           // [BM]

  const int tid = threadIdx.x, lane = tid & 31, wid = tid >> 5;
  const int wm = wid & 1, wn = wid >> 1;               // 2 x 4 wave grid -> 64 x 128 tile
  const int r = lane & 15, hf = lane >> 4;

  if (tid < BM) {
    bool valid = (mbase + tid < cnt);
    tok_lds[tid]  = valid ? tok_ids[(size_t)e * BATCH + mbase + tid] : 0;
    gate_lds[tid] = valid ? gates[(size_t)e * BATCH + mbase + tid] : 0.f;
  }
  __syncthreads();

  // hoist expert pointers into locals (keep in SGPRs, no kernarg reload in loops)
  const float *w1 = P.e[e].w1, *b1 = P.e[e].b1, *g1 = P.e[e].g1, *be1 = P.e[e].be1;
  const float *w2 = P.e[e].w2, *b2 = P.e[e].b2, *g2 = P.e[e].g2, *be2 = P.e[e].be2;

  // A-stage geometry (fixed per thread): row & 16B chunk of the gathered x row
  const int arow = tid >> 2, ach = tid & 3;
  const __bf16* a_src = xb + (size_t)tok_lds[arow] * INPUT_DIM + ach * 8;
  __bf16* a_dst0 = A_lds + arow * STG_LD + ach * 8;

  // ---------- phase 1: h = relu((x @ w1 + b1)*scale1 + be1) -> LDS ----------
  for (int n0 = 0; n0 < S; n0 += BN) {
    v8f c[2][2];
#pragma unroll
    for (int mi = 0; mi < 2; ++mi)
#pragma unroll
      for (int ni = 0; ni < 2; ++ni) c[mi][ni] = (v8f){0,0,0,0,0,0,0,0};

    // prologue: stage tile k0=0 into buffer 0
    {
      async_copy16(a_dst0, a_src);                     // A via async DMA
      float vv[16];
#pragma unroll
      for (int j = 0; j < 8; ++j) {                    // B: 16 coalesced dword loads
        int idx = tid + 256 * j, n = idx & 127, k = (idx >> 7) * 2;
        int gn = n0 + n;  int gcl = gn < S ? gn : 0;
        vv[2*j]   = w1[(size_t)(k)     * S + gcl];
        vv[2*j+1] = w1[(size_t)(k + 1) * S + gcl];
        if (gn >= S) { vv[2*j] = 0.f; vv[2*j+1] = 0.f; }
      }
#pragma unroll
      for (int j = 0; j < 8; ++j) {
        int idx = tid + 256 * j, n = idx & 127, k = (idx >> 7) * 2;
        *(unsigned*)(B_lds + n * STG_LD + k) = pack_bf16(vv[2*j], vv[2*j+1]);
      }
      wait_async0();
      __syncthreads();
    }

#pragma unroll 2
    for (int k0 = 0; k0 < INPUT_DIM; k0 += BK) {
      const int cur = (k0 >> 5) & 1, nxt = cur ^ 1;
      const bool have_next = (k0 + BK) < INPUT_DIM;
      float vv[16];
      if (have_next) {                                  // issue next-tile traffic first
        async_copy16(a_dst0 + nxt * A_BUF_ELEMS, a_src + k0 + BK);
#pragma unroll
        for (int j = 0; j < 8; ++j) {
          int idx = tid + 256 * j, n = idx & 127, k = (idx >> 7) * 2;
          int gn = n0 + n;  int gcl = gn < S ? gn : 0;
          vv[2*j]   = w1[(size_t)(k0 + BK + k)     * S + gcl];
          vv[2*j+1] = w1[(size_t)(k0 + BK + k + 1) * S + gcl];
          if (gn >= S) { vv[2*j] = 0.f; vv[2*j+1] = 0.f; }
        }
      }
      // compute current tile while next-tile loads are in flight
      const __bf16* Ab = A_lds + cur * A_BUF_ELEMS;
      const __bf16* Bb = B_lds + cur * B_BUF_ELEMS;
#pragma unroll
      for (int mi = 0; mi < 2; ++mi) {
        v16bf a = load_frag_a(Ab + (wm * 32 + mi * 16) * STG_LD, STG_LD, lane);
#pragma unroll
        for (int ni = 0; ni < 2; ++ni) {
          v16bf b = load_frag_b(Bb + (wn * 32 + ni * 16) * STG_LD, STG_LD, lane);
          c[mi][ni] = __builtin_amdgcn_wmma_f32_16x16x32_bf16(
              false, a, false, b, (short)0, c[mi][ni], false, false);
        }
      }
      if (have_next) {
#pragma unroll
        for (int j = 0; j < 8; ++j) {
          int idx = tid + 256 * j, n = idx & 127, k = (idx >> 7) * 2;
          *(unsigned*)(B_lds + nxt * B_BUF_ELEMS + n * STG_LD + k) = pack_bf16(vv[2*j], vv[2*j+1]);
        }
      }
      wait_async0();
      __syncthreads();
    }
    // epilogue 1: bias + eval-BN + relu -> bf16 h in LDS
#pragma unroll
    for (int mi = 0; mi < 2; ++mi)
#pragma unroll
      for (int ni = 0; ni < 2; ++ni) {
        int ncol = n0 + wn * 32 + ni * 16 + r;
        if (ncol < S) {
          float sc  = g1[ncol] * BN_RSQRT;
          float b1v = b1[ncol], bev = be1[ncol];
#pragma unroll
          for (int i = 0; i < 8; ++i) {
            int mrow = wm * 32 + mi * 16 + hf * 8 + i;   // C layout: M = i + half*8, N = r
            float hv = (c[mi][ni][i] + b1v) * sc + bev;
            h_lds[mrow * SH + ncol] = (__bf16)(hv > 0.f ? hv : 0.f);
          }
        }
      }
  }
  // zero K-padding columns [S, Sc)
  for (int idx = tid; idx < BM * (Sc - S); idx += 256) {
    int mrow = idx / (Sc - S), kc = S + idx % (Sc - S);
    h_lds[mrow * SH + kc] = (__bf16)0.f;
  }
  __syncthreads();

  // ---------- phase 2: y = (h @ w2 + b2)*scale2 + be2; out += gate * y ----------
  for (int n0 = 0; n0 < INPUT_DIM; n0 += BN) {
    v8f c[2][2];
#pragma unroll
    for (int mi = 0; mi < 2; ++mi)
#pragma unroll
      for (int ni = 0; ni < 2; ++ni) c[mi][ni] = (v8f){0,0,0,0,0,0,0,0};

    { // prologue: stage w2 tile k0=0 into buffer 0
      float vv[16];
#pragma unroll
      for (int j = 0; j < 8; ++j) {
        int idx = tid + 256 * j, n = idx & 127, k = (idx >> 7) * 2;
        int g0 = k < S ? k : 0, g1k = (k + 1) < S ? (k + 1) : 0;
        vv[2*j]   = w2[(size_t)g0  * INPUT_DIM + n0 + n];
        vv[2*j+1] = w2[(size_t)g1k * INPUT_DIM + n0 + n];
        if (k     >= S) vv[2*j]   = 0.f;
        if (k + 1 >= S) vv[2*j+1] = 0.f;
      }
#pragma unroll
      for (int j = 0; j < 8; ++j) {
        int idx = tid + 256 * j, n = idx & 127, k = (idx >> 7) * 2;
        *(unsigned*)(B_lds + n * STG_LD + k) = pack_bf16(vv[2*j], vv[2*j+1]);
      }
      __syncthreads();
    }

#pragma unroll 2
    for (int k0 = 0; k0 < Sc; k0 += BK) {
      const int cur = (k0 >> 5) & 1, nxt = cur ^ 1;
      const bool have_next = (k0 + BK) < Sc;
      float vv[16];
      if (have_next) {
#pragma unroll
        for (int j = 0; j < 8; ++j) {
          int idx = tid + 256 * j, n = idx & 127, k = k0 + BK + (idx >> 7) * 2;
          int g0 = k < S ? k : 0, g1k = (k + 1) < S ? (k + 1) : 0;
          vv[2*j]   = w2[(size_t)g0  * INPUT_DIM + n0 + n];
          vv[2*j+1] = w2[(size_t)g1k * INPUT_DIM + n0 + n];
          if (k     >= S) vv[2*j]   = 0.f;
          if (k + 1 >= S) vv[2*j+1] = 0.f;
        }
      }
      const __bf16* Bb = B_lds + cur * B_BUF_ELEMS;
#pragma unroll
      for (int mi = 0; mi < 2; ++mi) {
        v16bf a = load_frag_a(h_lds + (wm * 32 + mi * 16) * SH + k0, SH, lane);
#pragma unroll
        for (int ni = 0; ni < 2; ++ni) {
          v16bf b = load_frag_b(Bb + (wn * 32 + ni * 16) * STG_LD, STG_LD, lane);
          c[mi][ni] = __builtin_amdgcn_wmma_f32_16x16x32_bf16(
              false, a, false, b, (short)0, c[mi][ni], false, false);
        }
      }
      if (have_next) {
#pragma unroll
        for (int j = 0; j < 8; ++j) {
          int idx = tid + 256 * j, n = idx & 127, k = (idx >> 7) * 2;
          *(unsigned*)(B_lds + nxt * B_BUF_ELEMS + n * STG_LD + k) = pack_bf16(vv[2*j], vv[2*j+1]);
        }
      }
      __syncthreads();
    }
    // epilogue 2: scale + gate, atomic accumulate into residual-initialized out
#pragma unroll
    for (int mi = 0; mi < 2; ++mi)
#pragma unroll
      for (int ni = 0; ni < 2; ++ni) {
        int ncol = n0 + wn * 32 + ni * 16 + r;
        float sc2 = g2[ncol] * BN_RSQRT;
        float b2v = b2[ncol], be2v = be2[ncol];
#pragma unroll
        for (int i = 0; i < 8; ++i) {
          int mrow = wm * 32 + mi * 16 + hf * 8 + i;
          if (mbase + mrow < cnt) {
            float g  = gate_lds[mrow];
            float yv = ((c[mi][ni][i] + b2v) * sc2 + be2v) * g;
            atomicAdd(out + (size_t)tok_lds[mrow] * INPUT_DIM + ncol, yv);
          }
        }
      }
  }
}

// ================= host launcher =================
extern "C" void kernel_launch(void* const* d_in, const int* in_sizes, int n_in,
                              void* d_out, int out_size, void* d_ws, size_t ws_size,
                              hipStream_t stream) {
  (void)in_sizes; (void)n_in; (void)out_size; (void)ws_size;
  const float* x = (const float*)d_in[0];
  AllExperts P;
  for (int e = 0; e < NUM_EXPERTS; ++e) {
    const int b = 1 + e * 8;
    P.e[e].w1  = (const float*)d_in[b + 0];
    P.e[e].b1  = (const float*)d_in[b + 1];
    P.e[e].g1  = (const float*)d_in[b + 2];
    P.e[e].be1 = (const float*)d_in[b + 3];
    P.e[e].w2  = (const float*)d_in[b + 4];
    P.e[e].b2  = (const float*)d_in[b + 5];
    P.e[e].g2  = (const float*)d_in[b + 6];
    P.e[e].be2 = (const float*)d_in[b + 7];
  }
  const float* rw = (const float*)d_in[1 + NUM_EXPERTS * 8];
  const float* rb = (const float*)d_in[2 + NUM_EXPERTS * 8];
  float* out = (float*)d_out;

  // workspace carve (~9.6 MB total)
  char* w = (char*)d_ws;
  auto carve = [&](size_t bytes) { void* p = (void*)w; w += (bytes + 255) & ~(size_t)255; return p; };
  __bf16* xb     = (__bf16*)carve((size_t)BATCH * INPUT_DIM * sizeof(__bf16));
  float*  gdense = (float*) carve((size_t)BATCH * NUM_EXPERTS * sizeof(float));
  int*    tok    = (int*)   carve((size_t)NUM_EXPERTS * BATCH * sizeof(int));
  float*  gts    = (float*) carve((size_t)NUM_EXPERTS * BATCH * sizeof(float));
  int*    counts = (int*)   carve((size_t)NUM_EXPERTS * sizeof(int));

  moe_init_kernel<<<(BATCH * INPUT_DIM) / 256, 256, 0, stream>>>(x, out, xb);
  moe_router_kernel<<<BATCH / 8, 256, 0, stream>>>(x, rw, rb, gdense);
  moe_compact_kernel<<<NUM_EXPERTS, 256, 0, stream>>>(gdense, tok, gts, counts);
  moe_ffn_kernel<<<NUM_EXPERTS * (BATCH / BM), 256, FFN_SMEM, stream>>>(
      P, xb, tok, gts, counts, out);
}
